// GraphTransformer2Layer_37082747633995
// MI455X (gfx1250) — matrix operand order repaired and checked by reading, not compiled, not run on hardware
//
#include <hip/hip_runtime.h>
#include <hip/hip_bf16.h>
#include <math.h>
#include <stdint.h>

// ---------------------------------------------------------------------------
// Problem constants (from reference)
// ---------------------------------------------------------------------------
#define GN 50000
#define GE 800000
#define GK 128         // input dim / hidden dim between layers
#define GD1 128        // layer-1 stacked width (4 heads * 32)
#define GH1 4
#define GC1 32
#define GD2 10         // layer-2 width (1 head * 10)
#define GH2 1
#define GC2 10

typedef __attribute__((ext_vector_type(16))) _Float16 v16h;
typedef __attribute__((ext_vector_type(8)))  _Float16 v8h;
typedef __attribute__((ext_vector_type(8)))  float    v8f;

// ---------------------------------------------------------------------------
// Elementwise helpers
// ---------------------------------------------------------------------------
__global__ void k_f32_to_f16(const float* __restrict__ in, _Float16* __restrict__ out,
                             long long n) {
    long long i = (long long)blockIdx.x * blockDim.x + threadIdx.x;
    long long s = (long long)gridDim.x * blockDim.x;
    for (; i < n; i += s) out[i] = (_Float16)in[i];
}

__global__ void k_fill_f32(float* __restrict__ p, float v, long long n) {
    long long i = (long long)blockIdx.x * blockDim.x + threadIdx.x;
    long long s = (long long)gridDim.x * blockDim.x;
    for (; i < n; i += s) p[i] = v;
}

// Wt[n*K + k] = (f16) W[k*Nc + n]   (row-major [K][Nc] -> transposed [Nc][K])
__global__ void k_transpose_w(const float* __restrict__ W, _Float16* __restrict__ Wt,
                              int K, int Nc) {
    int idx = blockIdx.x * blockDim.x + threadIdx.x;
    if (idx >= K * Nc) return;
    int k = idx / Nc;
    int n = idx - k * Nc;
    Wt[(size_t)n * K + k] = (_Float16)W[idx];
}

// ---------------------------------------------------------------------------
// WMMA GEMM: C[M,Nc] = A16[M,K] * Bt16[Nc,K]^T + bias[Nc]
// one wave -> one 16x16 tile; blockDim.x = 32 * ceil(Nc/16); grid.x = M/16
// ---------------------------------------------------------------------------
__global__ void k_wmma_gemm(const _Float16* __restrict__ A,
                            const _Float16* __restrict__ Bt,
                            const float* __restrict__ bias,
                            float* __restrict__ C,
                            int M, int K, int Nc) {
    const int lane = threadIdx.x & 31;
    const int wave = threadIdx.x >> 5;
    const int rowBase = blockIdx.x * 16;
    const int nBase = wave * 16;

    const int mrow = rowBase + (lane & 15);
    const int mC = (mrow < M) ? mrow : (M - 1);
    const int n = nBase + (lane & 15);
    const int nC = (n < Nc) ? n : (Nc - 1);
    const int hi = lane >> 4;        // 0 or 1 (half-wave)
    const int kloA = hi * 8;         // A: K sub-chunk within 16
    const int kloB = hi * 8;         // B: K sub-chunk within 16 (same as A)

    const float bv = bias[nC];
    v8f c;
#pragma unroll
    for (int r = 0; r < 8; ++r) c[r] = bv;

    const _Float16* arow = A + (size_t)mC * K;
    const _Float16* brow = Bt + (size_t)nC * K;

    for (int kb = 0; kb < K; kb += 32) {
        // A fragment: a[0..7] = A[m][kb+kloA+0..7], a[8..15] = A[m][kb+16+kloA+0..7]
        v8h alo = *(const v8h*)(arow + kb + kloA);
        v8h ahi = *(const v8h*)(arow + kb + 16 + kloA);
        v16h a = __builtin_shufflevector(alo, ahi,
                                         0, 1, 2, 3, 4, 5, 6, 7,
                                         8, 9, 10, 11, 12, 13, 14, 15);
        // B fragment: b[0..7] = Bt[n][kb+kloB+0..7], b[8..15] = Bt[n][kb+16+kloB+0..7]
        v8h blo = *(const v8h*)(brow + kb + kloB);
        v8h bhi = *(const v8h*)(brow + kb + 16 + kloB);
        v16h b = __builtin_shufflevector(blo, bhi,
                                         0, 1, 2, 3, 4, 5, 6, 7,
                                         8, 9, 10, 11, 12, 13, 14, 15);
        // D = A*B + C  (emits v_wmma_f32_16x16x32_f16)
        c = __builtin_amdgcn_wmma_f32_16x16x32_f16(
                /*neg_a=*/false, a, /*neg_b=*/false, b,
                /*c_mod=*/(short)0, c, /*reuse_a=*/false, /*reuse_b=*/false);
    }

    if (n < Nc) {
#pragma unroll
        for (int r = 0; r < 8; ++r) {
            int m = rowBase + r + hi * 8;
            if (m < M) C[(size_t)m * Nc + n] = c[r];
        }
    }
}

// ---------------------------------------------------------------------------
// Edge phase
// ---------------------------------------------------------------------------
__device__ __forceinline__ void atomicMaxF32(float* addr, float val) {
    if (val >= 0.0f)
        atomicMax((int*)addr, __float_as_int(val));
    else
        atomicMin((unsigned int*)addr, __float_as_uint(val));
}

// logits[e*H+h] = scale * dot(q[dst[e],h,:], k[src[e],h,:]); atomicMax into mbuf
// VEC-wide gathers (float4 for layer1: head offsets are 16B aligned;
//                   float2 for layer2: rows are 40B -> 8B aligned)
template <int H, int C, int VEC>
__global__ void k_edge_logits(const float* __restrict__ q, const float* __restrict__ k,
                              const int* __restrict__ src, const int* __restrict__ dst,
                              float* __restrict__ logits, float* __restrict__ mbuf,
                              int E, float scale) {
    int id = blockIdx.x * blockDim.x + threadIdx.x;
    if (id >= E * H) return;
    int e = id / H;
    int h = id - e * H;
    int s = src[e], d = dst[e];
    const float* qp = q + (size_t)d * (H * C) + (size_t)h * C;
    const float* kp = k + (size_t)s * (H * C) + (size_t)h * C;
    float acc = 0.0f;
#pragma unroll
    for (int c = 0; c < C; c += VEC) {
        float qv[VEC], kv[VEC];
        if (VEC == 4) {
            *(float4*)qv = *(const float4*)(qp + c);
            *(float4*)kv = *(const float4*)(kp + c);
        } else {
            *(float2*)qv = *(const float2*)(qp + c);
            *(float2*)kv = *(const float2*)(kp + c);
        }
#pragma unroll
        for (int j = 0; j < VEC; ++j) acc = fmaf(qv[j], kv[j], acc);
    }
    acc *= scale;
    logits[id] = acc;
    atomicMaxF32(&mbuf[(size_t)d * H + h], acc);
}

// in-place: logits -> exp(logit - m[dst]); atomicAdd into denom
__global__ void k_edge_exp(float* __restrict__ logits, const float* __restrict__ mbuf,
                           float* __restrict__ denom, const int* __restrict__ dst,
                           int E, int H) {
    int id = blockIdx.x * blockDim.x + threadIdx.x;
    if (id >= E * H) return;
    int e = id / H;
    int h = id - e * H;
    int d = dst[e];
    float mm = mbuf[(size_t)d * H + h];
    if (!isfinite(mm)) mm = 0.0f;
    float ex = expf(logits[id] - mm);
    logits[id] = ex;
    atomicAdd(&denom[(size_t)d * H + h], ex);
}

// in-place: ew -> alpha = ew / (denom[dst] + eps)   (once per (e,h))
__global__ void k_edge_alpha(float* __restrict__ ew, const float* __restrict__ denom,
                             const int* __restrict__ dst, int E, int H) {
    int id = blockIdx.x * blockDim.x + threadIdx.x;
    if (id >= E * H) return;
    int e = id / H;
    int h = id - e * H;
    int d = dst[e];
    ew[id] = ew[id] / (denom[(size_t)d * H + h] + 1e-16f);
}

// agg[dst, t] += alpha[e,h] * v[src, t]; one thread handles VEC channels
template <int H, int C, int VEC>
__global__ void k_edge_agg(const float* __restrict__ alpha,
                           const float* __restrict__ v,
                           const int* __restrict__ src, const int* __restrict__ dst,
                           float* __restrict__ agg, int E) {
    const int D = H * C;
    const int G = D / VEC;                    // vector groups per edge
    long long total = (long long)E * G;
    long long i = (long long)blockIdx.x * blockDim.x + threadIdx.x;
    long long stride = (long long)gridDim.x * blockDim.x;
    for (; i < total; i += stride) {
        int e = (int)(i / G);
        int g = (int)(i - (long long)e * G);
        int t = g * VEC;
        int h = t / C;
        int s = src[e], d = dst[e];
        float al = alpha[(size_t)e * H + h];
        float vv[VEC];
        if (VEC == 4) *(float4*)vv = *(const float4*)(v + (size_t)s * D + t);
        else          *(float2*)vv = *(const float2*)(v + (size_t)s * D + t);
        float* ap = agg + (size_t)d * D + t;
#pragma unroll
        for (int j = 0; j < VEC; ++j) atomicAdd(ap + j, al * vv[j]);
    }
}

// h16 = f16( elu(agg + skip) )
__global__ void k_node_elu_h16(const float* __restrict__ agg, const float* __restrict__ skip,
                               _Float16* __restrict__ h16, long long n) {
    long long i = (long long)blockIdx.x * blockDim.x + threadIdx.x;
    long long s = (long long)gridDim.x * blockDim.x;
    for (; i < n; i += s) {
        float t = agg[i] + skip[i];
        float r = (t > 0.0f) ? t : expm1f(t);
        h16[i] = (_Float16)r;
    }
}

// out = agg2 + s2
__global__ void k_node_add(const float* __restrict__ agg, const float* __restrict__ skip,
                           float* __restrict__ out, long long n) {
    long long i = (long long)blockIdx.x * blockDim.x + threadIdx.x;
    long long s = (long long)gridDim.x * blockDim.x;
    for (; i < n; i += s) out[i] = agg[i] + skip[i];
}

// ---------------------------------------------------------------------------
// Launch
// ---------------------------------------------------------------------------
static inline int gs(long long n, int b) {
    long long g = (n + b - 1) / b;
    if (g > 1048576) g = 1048576;   // grid-stride kernels cap the grid
    return (int)g;
}

extern "C" void kernel_launch(void* const* d_in, const int* in_sizes, int n_in,
                              void* d_out, int out_size, void* d_ws, size_t ws_size,
                              hipStream_t stream) {
    (void)in_sizes; (void)n_in; (void)out_size; (void)ws_size;

    const float* x   = (const float*)d_in[0];
    const int*   ei  = (const int*)d_in[1];
    const int*   src = ei;          // edge_index[0]
    const int*   dst = ei + GE;     // edge_index[1]
    const float* Wq1 = (const float*)d_in[2];  const float* bq1 = (const float*)d_in[3];
    const float* Wk1 = (const float*)d_in[4];  const float* bk1 = (const float*)d_in[5];
    const float* Wv1 = (const float*)d_in[6];  const float* bv1 = (const float*)d_in[7];
    const float* Ws1 = (const float*)d_in[8];  const float* bs1 = (const float*)d_in[9];
    const float* Wq2 = (const float*)d_in[10]; const float* bq2 = (const float*)d_in[11];
    const float* Wk2 = (const float*)d_in[12]; const float* bk2 = (const float*)d_in[13];
    const float* Wv2 = (const float*)d_in[14]; const float* bv2 = (const float*)d_in[15];
    const float* Ws2 = (const float*)d_in[16]; const float* bs2 = (const float*)d_in[17];
    float* out = (float*)d_out;

    // ---- workspace bump allocator ----
    uint8_t* wsp = (uint8_t*)d_ws;
    size_t off = 0;
    auto alloc = [&](size_t bytes) -> void* {
        void* p = wsp + off;
        off = (off + bytes + 255) & ~(size_t)255;
        return p;
    };

    _Float16* x16  = (_Float16*)alloc((size_t)GN * GK * 2);
    _Float16* h16  = (_Float16*)alloc((size_t)GN * GK * 2);
    _Float16* wtq1 = (_Float16*)alloc((size_t)GD1 * GK * 2);
    _Float16* wtk1 = (_Float16*)alloc((size_t)GD1 * GK * 2);
    _Float16* wtv1 = (_Float16*)alloc((size_t)GD1 * GK * 2);
    _Float16* wts1 = (_Float16*)alloc((size_t)GD1 * GK * 2);
    _Float16* wtq2 = (_Float16*)alloc((size_t)GD2 * GK * 2);
    _Float16* wtk2 = (_Float16*)alloc((size_t)GD2 * GK * 2);
    _Float16* wtv2 = (_Float16*)alloc((size_t)GD2 * GK * 2);
    _Float16* wts2 = (_Float16*)alloc((size_t)GD2 * GK * 2);
    float* q1   = (float*)alloc((size_t)GN * GD1 * 4);
    float* kk1  = (float*)alloc((size_t)GN * GD1 * 4);
    float* v1   = (float*)alloc((size_t)GN * GD1 * 4);
    float* s1   = (float*)alloc((size_t)GN * GD1 * 4);
    float* agg1 = (float*)alloc((size_t)GN * GD1 * 4);
    float* lg1  = (float*)alloc((size_t)GE * GH1 * 4);   // logits -> exp -> alpha (in place)
    float* m1   = (float*)alloc((size_t)GN * GH1 * 4);
    float* dn1  = (float*)alloc((size_t)GN * GH1 * 4);
    float* q2   = (float*)alloc((size_t)GN * GD2 * 4);
    float* kk2  = (float*)alloc((size_t)GN * GD2 * 4);
    float* v2   = (float*)alloc((size_t)GN * GD2 * 4);
    float* s2   = (float*)alloc((size_t)GN * GD2 * 4);
    float* agg2 = (float*)alloc((size_t)GN * GD2 * 4);
    float* lg2  = (float*)alloc((size_t)GE * GH2 * 4);
    float* m2   = (float*)alloc((size_t)GN * GH2 * 4);
    float* dn2  = (float*)alloc((size_t)GN * GH2 * 4);

    const int B = 256;

    // ---- precision conversion & weight transpose ----
    k_f32_to_f16<<<gs((long long)GN * GK, B), B, 0, stream>>>(x, x16, (long long)GN * GK);
    k_transpose_w<<<gs(GK * GD1, B), B, 0, stream>>>(Wq1, wtq1, GK, GD1);
    k_transpose_w<<<gs(GK * GD1, B), B, 0, stream>>>(Wk1, wtk1, GK, GD1);
    k_transpose_w<<<gs(GK * GD1, B), B, 0, stream>>>(Wv1, wtv1, GK, GD1);
    k_transpose_w<<<gs(GK * GD1, B), B, 0, stream>>>(Ws1, wts1, GK, GD1);
    k_transpose_w<<<gs(GK * GD2, B), B, 0, stream>>>(Wq2, wtq2, GK, GD2);
    k_transpose_w<<<gs(GK * GD2, B), B, 0, stream>>>(Wk2, wtk2, GK, GD2);
    k_transpose_w<<<gs(GK * GD2, B), B, 0, stream>>>(Wv2, wtv2, GK, GD2);
    k_transpose_w<<<gs(GK * GD2, B), B, 0, stream>>>(Ws2, wts2, GK, GD2);

    // ---- layer 1 projections (WMMA) ----
    const int gemmGrid = GN / 16;               // 50000 = 3125*16, exact
    const dim3 blk1(32 * (GD1 / 16));           // 8 waves = 8 column tiles
    k_wmma_gemm<<<gemmGrid, blk1, 0, stream>>>(x16, wtq1, bq1, q1, GN, GK, GD1);
    k_wmma_gemm<<<gemmGrid, blk1, 0, stream>>>(x16, wtk1, bk1, kk1, GN, GK, GD1);
    k_wmma_gemm<<<gemmGrid, blk1, 0, stream>>>(x16, wtv1, bv1, v1, GN, GK, GD1);
    k_wmma_gemm<<<gemmGrid, blk1, 0, stream>>>(x16, wts1, bs1, s1, GN, GK, GD1);

    // ---- layer 1 edge softmax + aggregation ----
    k_fill_f32<<<gs((long long)GN * GH1, B), B, 0, stream>>>(m1, -INFINITY, (long long)GN * GH1);
    k_fill_f32<<<gs((long long)GN * GH1, B), B, 0, stream>>>(dn1, 0.0f, (long long)GN * GH1);
    k_fill_f32<<<gs((long long)GN * GD1, B), B, 0, stream>>>(agg1, 0.0f, (long long)GN * GD1);
    k_edge_logits<GH1, GC1, 4><<<(GE * GH1 + B - 1) / B, B, 0, stream>>>(
        q1, kk1, src, dst, lg1, m1, GE, rsqrtf((float)GC1));
    k_edge_exp<<<(GE * GH1 + B - 1) / B, B, 0, stream>>>(lg1, m1, dn1, dst, GE, GH1);
    k_edge_alpha<<<(GE * GH1 + B - 1) / B, B, 0, stream>>>(lg1, dn1, dst, GE, GH1);
    k_edge_agg<GH1, GC1, 4><<<gs((long long)GE * (GD1 / 4), B), B, 0, stream>>>(
        lg1, v1, src, dst, agg1, GE);

    // ---- ELU + skip, convert to f16 for layer-2 GEMMs ----
    k_node_elu_h16<<<gs((long long)GN * GD1, B), B, 0, stream>>>(
        agg1, s1, h16, (long long)GN * GD1);

    // ---- layer 2 projections (WMMA, Nc=10 -> 1 column tile) ----
    const dim3 blk2(32);
    k_wmma_gemm<<<gemmGrid, blk2, 0, stream>>>(h16, wtq2, bq2, q2, GN, GK, GD2);
    k_wmma_gemm<<<gemmGrid, blk2, 0, stream>>>(h16, wtk2, bk2, kk2, GN, GK, GD2);
    k_wmma_gemm<<<gemmGrid, blk2, 0, stream>>>(h16, wtv2, bv2, v2, GN, GK, GD2);
    k_wmma_gemm<<<gemmGrid, blk2, 0, stream>>>(h16, wts2, bs2, s2, GN, GK, GD2);

    // ---- layer 2 edge softmax + aggregation ----
    k_fill_f32<<<gs((long long)GN * GH2, B), B, 0, stream>>>(m2, -INFINITY, (long long)GN * GH2);
    k_fill_f32<<<gs((long long)GN * GH2, B), B, 0, stream>>>(dn2, 0.0f, (long long)GN * GH2);
    k_fill_f32<<<gs((long long)GN * GD2, B), B, 0, stream>>>(agg2, 0.0f, (long long)GN * GD2);
    k_edge_logits<GH2, GC2, 2><<<(GE * GH2 + B - 1) / B, B, 0, stream>>>(
        q2, kk2, src, dst, lg2, m2, GE, rsqrtf((float)GC2));
    k_edge_exp<<<(GE * GH2 + B - 1) / B, B, 0, stream>>>(lg2, m2, dn2, dst, GE, GH2);
    k_edge_alpha<<<(GE * GH2 + B - 1) / B, B, 0, stream>>>(lg2, dn2, dst, GE, GH2);
    k_edge_agg<GH2, GC2, 2><<<gs((long long)GE * (GD2 / 2), B), B, 0, stream>>>(
        lg2, v2, src, dst, agg2, GE);

    // ---- final: out = agg2 + h @ Ws2 + bs2 ----
    k_node_add<<<gs((long long)GN * GD2, B), B, 0, stream>>>(
        agg2, s2, out, (long long)GN * GD2);
}
